// NodeModel_50371376447950
// MI455X (gfx1250) — matrix operand order repaired and checked by reading, not compiled, not run on hardware
//
#include <hip/hip_runtime.h>
#include <hip/hip_bf16.h>

typedef __attribute__((ext_vector_type(16))) _Float16 v16h;
typedef __attribute__((ext_vector_type(8)))  float    v8f;

#define BN_EPS 1e-5f

// ---------------------------------------------------------------------------
// Fragment helpers (wave32 WMMA 16x16x32 f16 layouts per CDNA5 ISA 7.12.2)
// A 16x32 f16: lane l holds row M=l%16; halves [ (l/16)*8 .. +7 ] -> elems 0..7,
//              halves [ 16+(l/16)*8 .. +7 ] -> elems 8..15.
// B 32x16 f16 (from transposed weights Wt[o][k], row-major, ld=Kdim):
//              lane l holds col N=l%16; same K layout as A (8h+i, 16+8h+i).
// C/D 16x16 f32: lane l holds col N=l%16, rows M=(l/16)*8+r for elem r.
// ---------------------------------------------------------------------------

typedef __attribute__((ext_vector_type(8)))  _Float16 v8h;

__device__ __forceinline__ v16h bfrag(const _Float16* __restrict__ Wt, int Kdim,
                                      int n0, int k0, int lane) {
  const int n  = lane & 15;
  const int kh = (lane >> 4) * 8;
  const _Float16* p = Wt + (size_t)(n0 + n) * Kdim + k0 + kh;
  v8h lo = *(const v8h*)p;
  v8h hi = *(const v8h*)(p + 16);
  return __builtin_shufflevector(lo, hi, 0,1,2,3,4,5,6,7,8,9,10,11,12,13,14,15);
}

__device__ __forceinline__ v16h bfrag_lds(const _Float16* Ws, int Kdim,
                                          int n0, int k0, int lane) {
  const int n  = lane & 15;
  const int kh = (lane >> 4) * 8;
  const _Float16* p = Ws + (n0 + n) * Kdim + k0 + kh;
  v8h lo = *(const v8h*)p;
  v8h hi = *(const v8h*)(p + 16);
  return __builtin_shufflevector(lo, hi, 0,1,2,3,4,5,6,7,8,9,10,11,12,13,14,15);
}

__device__ __forceinline__ v16h afrag_f32(const float* __restrict__ row, int k0, int lane) {
  const int kh = (lane >> 4) * 8;
  const float* plo = row + k0 + kh;
  const float* phi = row + k0 + 16 + kh;
  v16h a;
#pragma unroll
  for (int i = 0; i < 8; ++i) {
    a[i]     = (_Float16)plo[i];
    a[8 + i] = (_Float16)phi[i];
  }
  return a;
}

// A from stored f16 activations with per-channel BN affine + ReLU applied on the fly.
__device__ __forceinline__ v16h afrag_h_bnrelu(const _Float16* __restrict__ row, int k0,
                                               int lane, const float* __restrict__ A,
                                               const float* __restrict__ C) {
  const int kh = (lane >> 4) * 8;
  v16h a;
#pragma unroll
  for (int i = 0; i < 8; ++i) {
    const int klo = k0 + kh + i;
    const int khi = k0 + 16 + kh + i;
    float vlo = fmaxf(fmaf((float)row[klo], A[klo], C[klo]), 0.0f);
    float vhi = fmaxf(fmaf((float)row[khi], A[khi], C[khi]), 0.0f);
    a[i]     = (_Float16)vlo;
    a[8 + i] = (_Float16)vhi;
  }
  return a;
}

// A slice for the aggregate block: agg = (cnt>0) ? a2*(seg/cnt)+c2 : 0
__device__ __forceinline__ v16h afrag_agg(const float* __restrict__ segrow, int k0,
                                          int lane, float inv, float cn,
                                          const float* __restrict__ A,
                                          const float* __restrict__ C) {
  const int kh = (lane >> 4) * 8;
  v16h a;
#pragma unroll
  for (int i = 0; i < 8; ++i) {
    const int klo = k0 + kh + i;
    const int khi = k0 + 16 + kh + i;
    float vlo = (cn > 0.0f) ? fmaf(segrow[klo] * inv, A[klo], C[klo]) : 0.0f;
    float vhi = (cn > 0.0f) ? fmaf(segrow[khi] * inv, A[khi], C[khi]) : 0.0f;
    a[i]     = (_Float16)vlo;
    a[8 + i] = (_Float16)vhi;
  }
  return a;
}

// Reduce per-lane BN stats (accumulated across the whole tile loop) into LDS,
// then (by caller) one global atomic per channel per block.
__device__ __forceinline__ void stats_reduce_to_lds(float* __restrict__ ls,
                                                    const float* __restrict__ ssum,
                                                    const float* __restrict__ ssq,
                                                    int lane) {
#pragma unroll
  for (int nt = 0; nt < 4; ++nt) {
    float s = ssum[nt] + __shfl_xor(ssum[nt], 16, 32);
    float q = ssq[nt]  + __shfl_xor(ssq[nt], 16, 32);
    if (lane < 16) {
      atomicAdd(&ls[nt * 16 + lane], s);
      atomicAdd(&ls[64 + nt * 16 + lane], q);
    }
  }
}

// ---------------------------------------------------------------------------
// Kernel 0: convert + transpose all four weights f32[K,64] -> f16[64,K]
// ---------------------------------------------------------------------------
__global__ void prep_weights(const float* __restrict__ W1a, const float* __restrict__ W1b,
                             const float* __restrict__ W2a, const float* __restrict__ W2b,
                             _Float16* __restrict__ Wt1a, _Float16* __restrict__ Wt1b,
                             _Float16* __restrict__ Wt2a, _Float16* __restrict__ Wt2b) {
  int i = blockIdx.x * blockDim.x + threadIdx.x;
  if (i < 8192) {                       // W1a: [128,64]
    int k = i >> 6, o = i & 63;
    Wt1a[o * 128 + k] = (_Float16)W1a[i];
  } else if (i < 12288) {               // W1b: [64,64]
    int j = i - 8192, k = j >> 6, o = j & 63;
    Wt1b[o * 64 + k] = (_Float16)W1b[j];
  } else if (i < 24576) {               // W2a: [192,64]
    int j = i - 12288, k = j >> 6, o = j & 63;
    Wt2a[o * 192 + k] = (_Float16)W2a[j];
  } else if (i < 28672) {               // W2b: [64,64]
    int j = i - 24576, k = j >> 6, o = j & 63;
    Wt2b[o * 64 + k] = (_Float16)W2b[j];
  }
}

__global__ void zero_f32(float* __restrict__ p, long long n) {
  long long i = (long long)blockIdx.x * blockDim.x + threadIdx.x;
  long long stride = (long long)gridDim.x * blockDim.x;
  for (; i < n; i += stride) p[i] = 0.0f;
}

// stats[0:64]=sum, stats[64:128]=sumsq over `count` rows. coef[0:64]=a, coef[64:128]=c
// so that bn(x) = a*x + c.
__global__ void finalize_bn(const float* __restrict__ stats, const float* __restrict__ g,
                            const float* __restrict__ b, float* __restrict__ coef,
                            float invCount) {
  int c = threadIdx.x;  // 64 threads
  float m = stats[c] * invCount;
  float v = stats[64 + c] * invCount - m * m;
  float a = g[c] * rsqrtf(v + BN_EPS);
  coef[c]      = a;
  coef[64 + c] = b[c] - m * a;
}

// ---------------------------------------------------------------------------
// Kernel 1: t1[e,:] = concat(x[col[e]], edge_attr[e]) @ W1a   (+ BN stats set0)
// Persistent: B fragments (W1a^T) pinned in VGPRs, grid-stride over 16-edge tiles.
// ---------------------------------------------------------------------------
__global__ void __launch_bounds__(256, 1)
edge_gemm1(const float* __restrict__ x, const float* __restrict__ ea,
           const int* __restrict__ ei, long long E,
           const _Float16* __restrict__ Wt1a,
           _Float16* __restrict__ t1, float* __restrict__ stats0) {
  __shared__ float ls[128];
  const int tid = threadIdx.x, lane = tid & 31, wid = tid >> 5;
  if (tid < 128) ls[tid] = 0.0f;
  __syncthreads();

  v16h bf[4][4];
#pragma unroll
  for (int ks = 0; ks < 4; ++ks)
#pragma unroll
    for (int nt = 0; nt < 4; ++nt) bf[ks][nt] = bfrag(Wt1a, 128, nt * 16, ks * 32, lane);

  float ssum[4] = {0.f, 0.f, 0.f, 0.f}, ssq[4] = {0.f, 0.f, 0.f, 0.f};
  const long long ntiles = E >> 4;
  const long long wstride = (long long)gridDim.x * 8;
  const int mb = (lane >> 4) * 8, n = lane & 15;

  for (long long tile = (long long)blockIdx.x * 8 + wid; tile < ntiles; tile += wstride) {
    const long long e0 = tile << 4;
    const long long e  = e0 + (lane & 15);
    const int ci = ei[E + e];  // destination-gather index (col)
    const float* xrow = x + (size_t)ci * 64;
    const float* erow = ea + (size_t)e * 64;
    v8f acc[4] = {};
#pragma unroll
    for (int ks = 0; ks < 4; ++ks) {
      const int k0 = ks * 32;
      v16h a = (k0 < 64) ? afrag_f32(xrow, k0, lane) : afrag_f32(erow, k0 - 64, lane);
#pragma unroll
      for (int nt = 0; nt < 4; ++nt)
        acc[nt] = __builtin_amdgcn_wmma_f32_16x16x32_f16(
            false, a, false, bf[ks][nt], (short)0, acc[nt], false, false);
    }
#pragma unroll
    for (int nt = 0; nt < 4; ++nt)
#pragma unroll
      for (int r = 0; r < 8; ++r) {
        float v = acc[nt][r];
        t1[(size_t)(e0 + mb + r) * 64 + nt * 16 + n] = (_Float16)v;
        ssum[nt] += v;
        ssq[nt]  += v * v;
      }
  }
  stats_reduce_to_lds(ls, ssum, ssq, lane);
  __syncthreads();
  if (tid < 128) atomicAdd(&stats0[tid], ls[tid]);
}

// ---------------------------------------------------------------------------
// Kernel 2: t2 = relu(bn1a(t1)) @ W1b ; scatter-add raw t2 to seg[row], cnt[row]
//           (+ BN stats set1 over edges; bn1b affine is applied post-mean)
// ---------------------------------------------------------------------------
__global__ void __launch_bounds__(256, 1)
edge_gemm2_scatter(const _Float16* __restrict__ t1,
                   const int* __restrict__ ei, long long E,
                   const _Float16* __restrict__ Wt1b,
                   const float* __restrict__ coef0,
                   float* __restrict__ seg, float* __restrict__ cnt,
                   float* __restrict__ stats1) {
  __shared__ float ls[128];
  const int tid = threadIdx.x, lane = tid & 31, wid = tid >> 5;
  if (tid < 128) ls[tid] = 0.0f;
  __syncthreads();

  v16h bf[2][4];
#pragma unroll
  for (int ks = 0; ks < 2; ++ks)
#pragma unroll
    for (int nt = 0; nt < 4; ++nt) bf[ks][nt] = bfrag(Wt1b, 64, nt * 16, ks * 32, lane);

  const float* A1 = coef0;
  const float* C1 = coef0 + 64;
  float ssum[4] = {0.f, 0.f, 0.f, 0.f}, ssq[4] = {0.f, 0.f, 0.f, 0.f};
  const long long ntiles = E >> 4;
  const long long wstride = (long long)gridDim.x * 8;
  const int mb = (lane >> 4) * 8, n = lane & 15;

  for (long long tile = (long long)blockIdx.x * 8 + wid; tile < ntiles; tile += wstride) {
    const long long e0 = tile << 4;
    const long long e  = e0 + (lane & 15);
    const _Float16* row = t1 + (size_t)e * 64;
    v8f acc[4] = {};
#pragma unroll
    for (int ks = 0; ks < 2; ++ks) {
      v16h a = afrag_h_bnrelu(row, ks * 32, lane, A1, C1);
#pragma unroll
      for (int nt = 0; nt < 4; ++nt)
        acc[nt] = __builtin_amdgcn_wmma_f32_16x16x32_f16(
            false, a, false, bf[ks][nt], (short)0, acc[nt], false, false);
    }
    int rw[8];
#pragma unroll
    for (int r = 0; r < 8; ++r) rw[r] = ei[e0 + mb + r];  // source row index
#pragma unroll
    for (int nt = 0; nt < 4; ++nt)
#pragma unroll
      for (int r = 0; r < 8; ++r) {
        float v = acc[nt][r];
        atomicAdd(&seg[(size_t)rw[r] * 64 + nt * 16 + n], v);
        ssum[nt] += v;
        ssq[nt]  += v * v;
      }
    if (lane < 16) atomicAdd(&cnt[ei[e0 + lane]], 1.0f);
  }
  stats_reduce_to_lds(ls, ssum, ssq, lane);
  __syncthreads();
  if (tid < 128) atomicAdd(&stats1[tid], ls[tid]);
}

// ---------------------------------------------------------------------------
// Kernel 3: t3 = concat(x, bn1b(seg/cnt), u[batch]) @ W2a   (+ stats set2)
// 24 B-fragments don't fit in VGPRs alongside the accumulators, so stage the
// transposed weight (24 KB) in LDS (320 KB/WGP) and feed WMMAs via ds_load_b128.
// ---------------------------------------------------------------------------
__global__ void node_gemm1(const float* __restrict__ x, const float* __restrict__ seg,
                           const float* __restrict__ cnt, const float* __restrict__ u,
                           const int* __restrict__ batch, int N,
                           const _Float16* __restrict__ Wt2a,
                           const float* __restrict__ coef1,
                           _Float16* __restrict__ t3, float* __restrict__ stats2) {
  __shared__ float ls[128];
  __shared__ _Float16 WtS[64 * 192];
  const int tid = threadIdx.x, lane = tid & 31, wid = tid >> 5;
  if (tid < 128) ls[tid] = 0.0f;
  {
    const uint32_t* src = (const uint32_t*)Wt2a;
    uint32_t* dst = (uint32_t*)WtS;
    for (int i = tid; i < 64 * 192 / 2; i += 256) dst[i] = src[i];
  }
  __syncthreads();

  const float* A2 = coef1;
  const float* C2 = coef1 + 64;
  float ssum[4] = {0.f, 0.f, 0.f, 0.f}, ssq[4] = {0.f, 0.f, 0.f, 0.f};
  const int ntiles = N >> 4;
  const int wstride = gridDim.x * 8;
  const int mb = (lane >> 4) * 8, n = lane & 15;

  for (int tile = blockIdx.x * 8 + wid; tile < ntiles; tile += wstride) {
    const int n0 = tile << 4;
    const int node = n0 + (lane & 15);
    const float* xrow   = x + (size_t)node * 64;
    const float* segrow = seg + (size_t)node * 64;
    const float* urow   = u + (size_t)batch[node] * 64;
    const float cn  = cnt[node];
    const float inv = (cn > 0.0f) ? (1.0f / cn) : 0.0f;
    v8f acc[4] = {};
#pragma unroll
    for (int ks = 0; ks < 6; ++ks) {
      const int k0 = ks * 32;
      v16h a;
      if (k0 < 64)       a = afrag_f32(xrow, k0, lane);
      else if (k0 < 128) a = afrag_agg(segrow, k0 - 64, lane, inv, cn, A2, C2);
      else               a = afrag_f32(urow, k0 - 128, lane);
#pragma unroll
      for (int nt = 0; nt < 4; ++nt)
        acc[nt] = __builtin_amdgcn_wmma_f32_16x16x32_f16(
            false, a, false, bfrag_lds(WtS, 192, nt * 16, k0, lane), (short)0, acc[nt],
            false, false);
    }
#pragma unroll
    for (int nt = 0; nt < 4; ++nt)
#pragma unroll
      for (int r = 0; r < 8; ++r) {
        float v = acc[nt][r];
        t3[(size_t)(n0 + mb + r) * 64 + nt * 16 + n] = (_Float16)v;
        ssum[nt] += v;
        ssq[nt]  += v * v;
      }
  }
  stats_reduce_to_lds(ls, ssum, ssq, lane);
  __syncthreads();
  if (tid < 128) atomicAdd(&stats2[tid], ls[tid]);
}

// ---------------------------------------------------------------------------
// Kernel 4: out_raw = relu(bn2a(t3)) @ W2b   (+ stats set3)
// ---------------------------------------------------------------------------
__global__ void __launch_bounds__(256, 1)
node_gemm2(const _Float16* __restrict__ t3, int N,
           const _Float16* __restrict__ Wt2b,
           const float* __restrict__ coef2,
           float* __restrict__ out, float* __restrict__ stats3) {
  __shared__ float ls[128];
  const int tid = threadIdx.x, lane = tid & 31, wid = tid >> 5;
  if (tid < 128) ls[tid] = 0.0f;
  __syncthreads();

  v16h bf[2][4];
#pragma unroll
  for (int ks = 0; ks < 2; ++ks)
#pragma unroll
    for (int nt = 0; nt < 4; ++nt) bf[ks][nt] = bfrag(Wt2b, 64, nt * 16, ks * 32, lane);

  const float* A3 = coef2;
  const float* C3 = coef2 + 64;
  float ssum[4] = {0.f, 0.f, 0.f, 0.f}, ssq[4] = {0.f, 0.f, 0.f, 0.f};
  const int ntiles = N >> 4;
  const int wstride = gridDim.x * 8;
  const int mb = (lane >> 4) * 8, n = lane & 15;

  for (int tile = blockIdx.x * 8 + wid; tile < ntiles; tile += wstride) {
    const int n0 = tile << 4;
    const int node = n0 + (lane & 15);
    const _Float16* row = t3 + (size_t)node * 64;
    v8f acc[4] = {};
#pragma unroll
    for (int ks = 0; ks < 2; ++ks) {
      v16h a = afrag_h_bnrelu(row, ks * 32, lane, A3, C3);
#pragma unroll
      for (int nt = 0; nt < 4; ++nt)
        acc[nt] = __builtin_amdgcn_wmma_f32_16x16x32_f16(
            false, a, false, bf[ks][nt], (short)0, acc[nt], false, false);
    }
#pragma unroll
    for (int nt = 0; nt < 4; ++nt)
#pragma unroll
      for (int r = 0; r < 8; ++r) {
        float v = acc[nt][r];
        out[(size_t)(n0 + mb + r) * 64 + nt * 16 + n] = v;
        ssum[nt] += v;
        ssq[nt]  += v * v;
      }
  }
  stats_reduce_to_lds(ls, ssum, ssq, lane);
  __syncthreads();
  if (tid < 128) atomicAdd(&stats3[tid], ls[tid]);
}

// Kernel 5: final BN applied in place on d_out.
__global__ void apply_bn_final(float* __restrict__ out, const float* __restrict__ coef,
                               long long total) {
  long long i = (long long)blockIdx.x * blockDim.x + threadIdx.x;
  long long stride = (long long)gridDim.x * blockDim.x;
  for (; i < total; i += stride) {
    int c = (int)(i & 63);
    out[i] = fmaf(out[i], coef[c], coef[64 + c]);
  }
}

// ---------------------------------------------------------------------------
extern "C" void kernel_launch(void* const* d_in, const int* in_sizes, int n_in,
                              void* d_out, int out_size, void* d_ws, size_t ws_size,
                              hipStream_t stream) {
  const float* x     = (const float*)d_in[0];
  const int*   ei    = (const int*)d_in[1];
  const float* ea    = (const float*)d_in[2];
  const float* u     = (const float*)d_in[3];
  const int*   batch = (const int*)d_in[4];
  const float* W1a = (const float*)d_in[5];
  const float* g1a = (const float*)d_in[6];
  const float* b1a = (const float*)d_in[7];
  const float* W1b = (const float*)d_in[8];
  const float* g1b = (const float*)d_in[9];
  const float* b1b = (const float*)d_in[10];
  const float* W2a = (const float*)d_in[11];
  const float* g2a = (const float*)d_in[12];
  const float* b2a = (const float*)d_in[13];
  const float* W2b = (const float*)d_in[14];
  const float* g2b = (const float*)d_in[15];
  const float* b2b = (const float*)d_in[16];

  const long long N = in_sizes[0] / 64;
  const long long E = in_sizes[1] / 2;

  char* ws = (char*)d_ws;
  auto align256 = [](size_t o) { return (o + 255) & ~(size_t)255; };
  size_t off = 0;
  _Float16* Wt1a = (_Float16*)(ws + off); off = align256(off + 64 * 128 * 2);
  _Float16* Wt1b = (_Float16*)(ws + off); off = align256(off + 64 * 64 * 2);
  _Float16* Wt2a = (_Float16*)(ws + off); off = align256(off + 64 * 192 * 2);
  _Float16* Wt2b = (_Float16*)(ws + off); off = align256(off + 64 * 64 * 2);
  size_t zeroStart = off;
  float* stats = (float*)(ws + off); off = align256(off + 4 * 128 * 4);  // 4 sets
  float* coef  = (float*)(ws + off); off = align256(off + 4 * 128 * 4);  // 4 sets
  float* cnt   = (float*)(ws + off); off = align256(off + (size_t)N * 4);
  float* seg   = (float*)(ws + off); off = align256(off + (size_t)N * 64 * 4);
  size_t zeroEnd = off;
  _Float16* t1 = (_Float16*)(ws + off); off = align256(off + (size_t)E * 64 * 2);
  _Float16* t3 = (_Float16*)(ws + off); off = align256(off + (size_t)N * 64 * 2);
  (void)ws_size; (void)n_in;

  // 0) Weight prep + zero accumulators
  prep_weights<<<112, 256, 0, stream>>>(W1a, W1b, W2a, W2b, Wt1a, Wt1b, Wt2a, Wt2b);
  const long long zf = (long long)((zeroEnd - zeroStart) / 4);
  zero_f32<<<4096, 256, 0, stream>>>((float*)(ws + zeroStart), zf);

  const long long edgeTiles = E >> 4;
  const long long nodeTiles = N >> 4;
  int edgeBlocks = (int)((edgeTiles + 7) / 8); if (edgeBlocks > 1024) edgeBlocks = 1024;
  int nodeBlocks = (int)((nodeTiles + 7) / 8); if (nodeBlocks > 512)  nodeBlocks = 512;

  // 1) Edge MLP layer 1 + stats
  edge_gemm1<<<edgeBlocks, 256, 0, stream>>>(x, ea, ei, E, Wt1a, t1, stats + 0 * 128);
  finalize_bn<<<1, 64, 0, stream>>>(stats + 0 * 128, g1a, b1a, coef + 0 * 128,
                                    1.0f / (float)E);
  // 2) Edge MLP layer 2 + scatter-mean accumulation + stats
  edge_gemm2_scatter<<<edgeBlocks, 256, 0, stream>>>(t1, ei, E, Wt1b, coef + 0 * 128,
                                                     seg, cnt, stats + 1 * 128);
  finalize_bn<<<1, 64, 0, stream>>>(stats + 1 * 128, g1b, b1b, coef + 1 * 128,
                                    1.0f / (float)E);
  // 3) Node MLP layer 1 + stats
  node_gemm1<<<nodeBlocks, 256, 0, stream>>>(x, seg, cnt, u, batch, (int)N, Wt2a,
                                             coef + 1 * 128, t3, stats + 2 * 128);
  finalize_bn<<<1, 64, 0, stream>>>(stats + 2 * 128, g2a, b2a, coef + 2 * 128,
                                    1.0f / (float)N);
  // 4) Node MLP layer 2 (raw to d_out) + stats
  node_gemm2<<<nodeBlocks, 256, 0, stream>>>(t3, (int)N, Wt2b, coef + 2 * 128,
                                             (float*)d_out, stats + 3 * 128);
  finalize_bn<<<1, 64, 0, stream>>>(stats + 3 * 128, g2b, b2b, coef + 3 * 128,
                                    1.0f / (float)N);
  // 5) Final BN in place
  apply_bn_final<<<2048, 256, 0, stream>>>((float*)d_out, coef + 3 * 128,
                                           (long long)N * 64);
  (void)out_size;
}
